// TriangleMultiplicativeUpdate_77807627534827
// MI455X (gfx1250) — matrix operand order repaired and checked by reading, not compiled, not run on hardware
//
#include <hip/hip_runtime.h>

// Triangle multiplicative update, N=512, C=128, for gfx1250 (MI455X).
// bf16 WMMA (v_wmma_f32_16x16x32_bf16), fp32 accumulation.
//   A fragment: row-major [M][K] LDS tile, lane reads K-contiguous (2x ds_load_b128).
//   B fragment: TRANSPOSED [N][K] LDS tile, lane reads K-contiguous (2x ds_load_b128).
// phase1: layernorm + 6 projections -> channel-major bf16 planes in d_ws
// phase2: 128 batched 512^3 GEMM pairs, tiles staged by the Tensor Data Mover
//         (tensor_load_to_lds, double-buffered, TENSORcnt-synchronized),
//         gate-combine -> tri (pixel-major bf16)
// phase3: tri @ w_out^T + b_out -> fp32 out
// Workspace: 7 buffers of N*N*C bf16 = 448 MiB.

#define N_DIM 512
#define C_DIM 128
#define NN (N_DIM * N_DIM)

typedef __bf16 bf16;
typedef __attribute__((ext_vector_type(16))) __bf16 bf16x16;
typedef __attribute__((ext_vector_type(8))) float floatx8;
typedef __attribute__((ext_vector_type(4))) unsigned int uint4v;
typedef __attribute__((ext_vector_type(4))) unsigned int u32x4;
typedef __attribute__((ext_vector_type(4))) int i32x4;
typedef __attribute__((ext_vector_type(8))) int i32x8;

struct Bits256 { uint4v lo, hi; };

__device__ __forceinline__ unsigned short f2bf(float f) {
  unsigned u = __builtin_bit_cast(unsigned, f);
  unsigned r = u + 0x7FFFu + ((u >> 16) & 1u);   // round-to-nearest-even
  return (unsigned short)(r >> 16);
}
__device__ __forceinline__ float bf2f(unsigned short h) {
  unsigned u = ((unsigned)h) << 16;
  return __builtin_bit_cast(float, u);
}

// Two aligned 16-byte LDS loads -> one 32-byte WMMA operand.
__device__ __forceinline__ bf16x16 load_frag16(const unsigned short* p0,
                                               const unsigned short* p1) {
  Bits256 t;
  t.lo = *(const uint4v*)p0;
  t.hi = *(const uint4v*)p1;
  return __builtin_bit_cast(bf16x16, t);
}

// A fragment (16x32 bf16, M x K) from row-major [M][K] LDS tile.
__device__ __forceinline__ bf16x16 load_a_frag(const unsigned short* t, int lda,
                                               int mBase, int kBase, int lane) {
  const unsigned short* row = t + (mBase + (lane & 15)) * lda + kBase + ((lane >> 4) << 3);
  return load_frag16(row, row + 16);
}

// B fragment (32x16 bf16, K x N) from TRANSPOSED [N][K] LDS tile.
__device__ __forceinline__ bf16x16 load_bT_frag(const unsigned short* t, int ldk,
                                                int nBase, int kBase, int lane) {
  const unsigned short* p = t + (nBase + (lane & 15)) * ldk + kBase + ((lane >> 4) << 3);
  return load_frag16(p, p + 16);
}

// ------------------------------------------------------------ TDM (phase 2)
// Issue a 2D tile DMA: `rows` rows of 32 bf16, row stride 512 elements,
// into LDS at byte offset lds_off. LDS padding: +4 DWORDs every 16 DWORDs
// stored -> LDS row pitch 80 B = 40 halfwords (16-B aligned for b128 frags).
__device__ __forceinline__ void tdm_load_tile(unsigned lds_off,
                                              const unsigned short* gptr,
                                              int rows) {
  unsigned long long ga = (unsigned long long)(__UINTPTR_TYPE__)gptr;
  u32x4 g0;
  g0[0] = 1u;                                            // count=1, user descriptor
  g0[1] = lds_off;                                       // lds_addr (bytes)
  g0[2] = (unsigned)(ga & 0xFFFFFFFFull);                // global_addr[31:0]
  g0[3] = (unsigned)((ga >> 32) & 0x1FFFFFFull)          // global_addr[56:32]
          | (2u << 30);                                  // type = 2 (image)
  i32x8 g1;
  g1[0] = (1 << 16)            // data_size = 2 bytes
        | (1 << 20)            // pad_enable
        | (3 << 22)            // pad_interval: every 16 DWORDs (one 64-B row)
        | (3 << 25);           // pad_amount: 4 DWORDs (16 B)
  g1[1] = (N_DIM & 0xFFFF) << 16;                        // tensor_dim0 lo (512)
  g1[2] = ((N_DIM & 0xFFFF) << 16);                      // dim0 hi=0 | tensor_dim1 lo (512)
  g1[3] = (32 << 16);                                    // dim1 hi=0 | tile_dim0 = 32
  g1[4] = rows & 0xFFFF;                                 // tile_dim1 = rows, tile_dim2 = 0
  g1[5] = N_DIM;                                         // tensor_dim0_stride lo = 512
  g1[6] = 0;                                             // stride hi | dim1_stride lo
  g1[7] = 0;
  i32x4 z4 = { 0, 0, 0, 0 };
#if __has_include(<hip/amd_detail/amd_gfx1250_TDM.h>)
  i32x8 z8 = { 0, 0, 0, 0, 0, 0, 0, 0 };
  __builtin_amdgcn_tensor_load_to_lds(g0, g1, z4, z4, z8, 0);
#else
  __builtin_amdgcn_tensor_load_to_lds(g0, g1, z4, z4, 0);
#endif
}

// ---------------------------------------------------------------- phase 1
// Per block: 64 consecutive pixels. LayerNorm -> 6 WMMA projections.
// Output planes (bf16, [c] major, NN each). Pixel (r,s), channel c:
//   buf0 aOut  [c][i=r][k=s]  (tr=0)   A of tri_out GEMM
//   buf1 bOutT [c][j=s][k=r]  (tr=1)   B^T of tri_out GEMM (b_o[k,j,c], j-major)
//   buf2 aInT  [c][i=s][k=r]  (tr=1)   A of tri_in GEMM  (a_in[k,i,c])
//   buf3 biT   [c][j=r][k=s]  (tr=0)   B^T of tri_in GEMM (b_i[j,k,c], j-major)
//   buf4 gOut  [c][r][s]      (tr=0)   sigmoid gate
//   buf5 gIn   [c][r][s]      (tr=0)   sigmoid gate
__global__ __launch_bounds__(256) void tri_phase1(
    const float* __restrict__ pair, const float* __restrict__ ln_w, const float* __restrict__ ln_b,
    const float* __restrict__ w_a_out, const float* __restrict__ w_b_out,
    const float* __restrict__ w_g_out, const float* __restrict__ b_g_out,
    const float* __restrict__ w_a_in, const float* __restrict__ w_b_in,
    const float* __restrict__ w_g_in, const float* __restrict__ b_g_in,
    unsigned short* __restrict__ ws) {
  __shared__ __align__(16) unsigned short xn[64][136];   // normalized x, bf16
  __shared__ __align__(16) unsigned short wt[128][136];  // weight, [n][k]

  const int tid = threadIdx.x;
  const int rowBase = blockIdx.x * 64;

  if (tid < 64) {
    const float* xp = pair + (size_t)(rowBase + tid) * C_DIM;
    float mu = 0.f;
#pragma unroll 8
    for (int k = 0; k < C_DIM; ++k) mu += xp[k];
    mu *= (1.0f / C_DIM);
    float var = 0.f;
#pragma unroll 8
    for (int k = 0; k < C_DIM; ++k) { float d = xp[k] - mu; var += d * d; }
    float inv = rsqrtf(var * (1.0f / C_DIM) + 1e-5f);
    for (int k = 0; k < C_DIM; ++k) {
      float v = (xp[k] - mu) * inv * ln_w[k] + ln_b[k];
      xn[tid][k] = f2bf(v);
    }
  }
  __syncthreads();

  const size_t BUF = (size_t)NN * C_DIM;
  unsigned short* dsts[6] = { ws, ws + BUF, ws + 2 * BUF, ws + 3 * BUF, ws + 4 * BUF, ws + 5 * BUF };
  const float* wsrc[6] = { w_a_out, w_b_out, w_a_in, w_b_in, w_g_out, w_g_in };
  const float* bsrc[6] = { nullptr, nullptr, nullptr, nullptr, b_g_out, b_g_in };
  const int tr[6] = { 0, 1, 1, 0, 0, 0 };

  const int wave = tid >> 5, lane = tid & 31;

#pragma unroll
  for (int p = 0; p < 6; ++p) {
    for (int i = tid; i < C_DIM * C_DIM; i += 256) {
      wt[i >> 7][i & 127] = f2bf(wsrc[p][i]);
    }
    __syncthreads();

#pragma unroll
    for (int t = 0; t < 4; ++t) {
      int tile = wave * 4 + t;      // 32 tiles: 4 (M) x 8 (N)
      int mt = tile >> 3, nt = tile & 7;
      floatx8 acc = {};
#pragma unroll
      for (int kk = 0; kk < C_DIM; kk += 32) {
        bf16x16 a = load_a_frag(&xn[0][0], 136, mt * 16, kk, lane);
        bf16x16 b = load_bT_frag(&wt[0][0], 136, nt * 16, kk, lane);
        acc = __builtin_amdgcn_wmma_f32_16x16x32_bf16(false, a, false, b,
                                                      (short)0, acc, false, false);
      }
      int half = lane >> 4;
      int c = nt * 16 + (lane & 15);
      float bias = bsrc[p] ? bsrc[p][c] : 0.f;
#pragma unroll
      for (int e = 0; e < 8; ++e) {
        int m = mt * 16 + half * 8 + e;
        int pix = rowBase + m;
        int r = pix >> 9, s = pix & (N_DIM - 1);
        float v = acc[e];
        if (bsrc[p]) v = 1.f / (1.f + expf(-(v + bias)));
        size_t off = (size_t)c * NN +
                     (tr[p] ? ((size_t)s * N_DIM + r) : ((size_t)r * N_DIM + s));
        dsts[p][off] = f2bf(v);
      }
    }
    __syncthreads();
  }
}

// ---------------------------------------------------------------- phase 2
// One block = one channel c + one 128x64 output tile.
//   triOut = aOut[c] @ bOutT[c]^T ; triIn = aInT[c] @ biT[c]^T
//   tri[i][j][c] = (gOut*triOut + gIn*triIn) / N   (bf16, pixel-major)
// Tiles staged by the TDM, double-buffered; wave 0 issues the DMAs and the
// TENSORcnt wait, one workgroup barrier per chunk publishes completion.
#define P2_A_H (128 * 40)            // A tile halfwords (incl. TDM pad)
#define P2_B_H (64 * 40)             // B tile halfwords
#define P2_BUF_H (2 * P2_A_H + 2 * P2_B_H)   // one buffer: A1 A2 B1 B2
__global__ __launch_bounds__(256) void tri_phase2(const unsigned short* __restrict__ ws,
                                                  unsigned short* __restrict__ tri) {
  extern __shared__ __align__(16) unsigned short smem[];  // 2 * P2_BUF_H halfwords

  const size_t BUF = (size_t)NN * C_DIM;
  const int c = blockIdx.y;
  const int bi = (int)(blockIdx.x >> 3) * 128;   // 4 row groups
  const int bj = (int)(blockIdx.x & 7) * 64;     // 8 col groups
  const unsigned short* pA1 = ws + 0 * BUF + (size_t)c * NN;  // aOut  [i][k]
  const unsigned short* pB1 = ws + 1 * BUF + (size_t)c * NN;  // bOutT [j][k]
  const unsigned short* pA2 = ws + 2 * BUF + (size_t)c * NN;  // aInT  [i][k]
  const unsigned short* pB2 = ws + 3 * BUF + (size_t)c * NN;  // biT   [j][k]
  const unsigned short* pGO = ws + 4 * BUF + (size_t)c * NN;
  const unsigned short* pGI = ws + 5 * BUF + (size_t)c * NN;

  const int tid = threadIdx.x, wave = tid >> 5, lane = tid & 31;
  const int mrow = wave * 16;   // 8 waves cover 128 rows

  floatx8 acc1[4] = {};
  floatx8 acc2[4] = {};

  // Prologue: DMA chunk 0 into buffer 0.
  if (wave == 0) {
    tdm_load_tile(0u * 2,                       pA1 + (size_t)bi * N_DIM, 128);
    tdm_load_tile((unsigned)P2_A_H * 2,         pA2 + (size_t)bi * N_DIM, 128);
    tdm_load_tile((unsigned)(2 * P2_A_H) * 2,   pB1 + (size_t)bj * N_DIM, 64);
    tdm_load_tile((unsigned)(2 * P2_A_H + P2_B_H) * 2, pB2 + (size_t)bj * N_DIM, 64);
  }

  for (int chunk = 0; chunk < N_DIM / 32; ++chunk) {
    if (wave == 0) __builtin_amdgcn_s_wait_tensorcnt((short)0);
    __syncthreads();   // all waves: current buffer ready, previous buffer free

    const int cur = chunk & 1;
    const int nxt = cur ^ 1;
    if (wave == 0 && chunk + 1 < N_DIM / 32) {
      int kk = (chunk + 1) * 32;
      unsigned base = (unsigned)(nxt * P2_BUF_H) * 2;
      tdm_load_tile(base,                              pA1 + (size_t)bi * N_DIM + kk, 128);
      tdm_load_tile(base + (unsigned)P2_A_H * 2,       pA2 + (size_t)bi * N_DIM + kk, 128);
      tdm_load_tile(base + (unsigned)(2 * P2_A_H) * 2, pB1 + (size_t)bj * N_DIM + kk, 64);
      tdm_load_tile(base + (unsigned)(2 * P2_A_H + P2_B_H) * 2,
                    pB2 + (size_t)bj * N_DIM + kk, 64);
    }

    const unsigned short* sA1 = smem + cur * P2_BUF_H;
    const unsigned short* sA2 = sA1 + P2_A_H;
    const unsigned short* sB1 = sA1 + 2 * P2_A_H;
    const unsigned short* sB2 = sB1 + P2_B_H;

    bf16x16 a1 = load_a_frag(sA1, 40, mrow, 0, lane);
    bf16x16 a2 = load_a_frag(sA2, 40, mrow, 0, lane);
#pragma unroll
    for (int n = 0; n < 4; ++n) {
      bf16x16 b1 = load_bT_frag(sB1, 40, n * 16, 0, lane);
      bf16x16 b2 = load_bT_frag(sB2, 40, n * 16, 0, lane);
      acc1[n] = __builtin_amdgcn_wmma_f32_16x16x32_bf16(false, a1, false, b1,
                                                        (short)0, acc1[n], false, false);
      acc2[n] = __builtin_amdgcn_wmma_f32_16x16x32_bf16(false, a2, false, b2,
                                                        (short)0, acc2[n], false, false);
    }
    __syncthreads();   // compute done before this buffer is overwritten
  }

  const float scale = 1.0f / (float)N_DIM;
  const int half = lane >> 4, nl = lane & 15;
#pragma unroll
  for (int n = 0; n < 4; ++n) {
#pragma unroll
    for (int e = 0; e < 8; ++e) {
      int i = bi + mrow + half * 8 + e;
      int j = bj + n * 16 + nl;
      size_t gidx = (size_t)i * N_DIM + j;
      float go = bf2f(pGO[gidx]);
      float gi = bf2f(pGI[gidx]);
      float t = (go * acc1[n][e] + gi * acc2[n][e]) * scale;
      tri[gidx * C_DIM + c] = f2bf(t);
    }
  }
}

// ---------------------------------------------------------------- phase 3
// out = tri @ w_out^T + b_out   (fp32 output, coalesced)
__global__ __launch_bounds__(256) void tri_phase3(const unsigned short* __restrict__ tri,
                                                  const float* __restrict__ w_out,
                                                  const float* __restrict__ b_out,
                                                  float* __restrict__ out) {
  __shared__ __align__(16) unsigned short At[64][136];   // [m][k]
  __shared__ __align__(16) unsigned short Wt[128][136];  // [n][k]
  const int tid = threadIdx.x;
  const int rowBase = blockIdx.x * 64;

  for (int i = tid; i < C_DIM * C_DIM; i += 256) {
    Wt[i >> 7][i & 127] = f2bf(w_out[i]);
  }
  for (int i = tid; i < 64 * C_DIM; i += 256) {
    int r = i >> 7, k = i & 127;
    At[r][k] = tri[(size_t)(rowBase + r) * C_DIM + k];
  }
  __syncthreads();

  const int wave = tid >> 5, lane = tid & 31;
#pragma unroll
  for (int t = 0; t < 4; ++t) {
    int tile = wave * 4 + t;
    int mt = tile >> 3, nt = tile & 7;
    floatx8 acc = {};
#pragma unroll
    for (int kk = 0; kk < C_DIM; kk += 32) {
      bf16x16 a = load_a_frag(&At[0][0], 136, mt * 16, kk, lane);
      bf16x16 b = load_bT_frag(&Wt[0][0], 136, nt * 16, kk, lane);
      acc = __builtin_amdgcn_wmma_f32_16x16x32_bf16(false, a, false, b,
                                                    (short)0, acc, false, false);
    }
    int half = lane >> 4;
    int c = nt * 16 + (lane & 15);
    float bias = b_out[c];
#pragma unroll
    for (int e = 0; e < 8; ++e) {
      int m = mt * 16 + half * 8 + e;
      out[(size_t)(rowBase + m) * C_DIM + c] = acc[e] + bias;
    }
  }
}

extern "C" void kernel_launch(void* const* d_in, const int* in_sizes, int n_in,
                              void* d_out, int out_size, void* d_ws, size_t ws_size,
                              hipStream_t stream) {
  const float* pair    = (const float*)d_in[0];
  const float* ln_w    = (const float*)d_in[1];
  const float* ln_b    = (const float*)d_in[2];
  const float* w_a_out = (const float*)d_in[3];
  const float* w_b_out = (const float*)d_in[4];
  const float* w_g_out = (const float*)d_in[5];
  const float* b_g_out = (const float*)d_in[6];
  const float* w_a_in  = (const float*)d_in[7];
  const float* w_b_in  = (const float*)d_in[8];
  const float* w_g_in  = (const float*)d_in[9];
  const float* b_g_in  = (const float*)d_in[10];
  const float* w_out   = (const float*)d_in[11];
  const float* b_out   = (const float*)d_in[12];

  unsigned short* ws = (unsigned short*)d_ws;   // 7 * NN * C bf16 = 448 MiB needed
  const size_t BUF = (size_t)NN * C_DIM;

  tri_phase1<<<NN / 64, 256, 0, stream>>>(pair, ln_w, ln_b,
                                          w_a_out, w_b_out, w_g_out, b_g_out,
                                          w_a_in, w_b_in, w_g_in, b_g_in, ws);
  tri_phase2<<<dim3(32, 128), 256, 2 * P2_BUF_H * sizeof(unsigned short), stream>>>(
      ws, ws + 6 * BUF);
  tri_phase3<<<NN / 64, 256, 0, stream>>>(ws + 6 * BUF, w_out, b_out, (float*)d_out);
}
